// APPNPNet_24300924961369
// MI455X (gfx1250) — compile-verified, matrix operand order checked by reading, not run on hardware
//
#include <hip/hip_runtime.h>
#include <hip/hip_bf16.h>
#include <math.h>

// ---------------- CDNA5 WMMA types ----------------
typedef __attribute__((ext_vector_type(16))) __bf16 v16bf;
typedef __attribute__((ext_vector_type(8)))  __bf16 v8bf;
typedef __attribute__((ext_vector_type(8)))  float  v8f;

#define N_FEAT 512
#define N_HID  256
#define N_CLS  64
#define K_HOPS 10
#define ALPHA  0.1f

// 16x32 bf16 fragment from f32 memory (A operand of GEMM1).
// p = row_base + hi*8 + kb*32 ; low half p[0..7], high half p[16..23]; 16B aligned.
__device__ __forceinline__ v16bf frag_from_f32(const float* __restrict__ p) {
    const float4* q = (const float4*)p;
    float4 a0 = q[0], a1 = q[1];   // elements 0..7
    float4 b0 = q[4], b1 = q[5];   // elements 16..23
    v16bf r;
    r[0]  = (__bf16)a0.x; r[1]  = (__bf16)a0.y; r[2]  = (__bf16)a0.z; r[3]  = (__bf16)a0.w;
    r[4]  = (__bf16)a1.x; r[5]  = (__bf16)a1.y; r[6]  = (__bf16)a1.z; r[7]  = (__bf16)a1.w;
    r[8]  = (__bf16)b0.x; r[9]  = (__bf16)b0.y; r[10] = (__bf16)b0.z; r[11] = (__bf16)b0.w;
    r[12] = (__bf16)b1.x; r[13] = (__bf16)b1.y; r[14] = (__bf16)b1.z; r[15] = (__bf16)b1.w;
    return r;
}

// Fragment from bf16 memory: two 16B loads, zero VALU conversion.
__device__ __forceinline__ v16bf frag_from_bf16(const __bf16* __restrict__ p) {
    v8bf a = *(const v8bf*)p;
    v8bf b = *(const v8bf*)(p + 16);
    v16bf r;
#pragma unroll
    for (int i = 0; i < 8; ++i) { r[i] = a[i]; r[8 + i] = b[i]; }
    return r;
}

// ---------------- weight f32 -> bf16 pre-conversion ----------------
__global__ void cvt_bf16(const float* __restrict__ in, __bf16* __restrict__ out, int n) {
    int i = blockIdx.x * blockDim.x + threadIdx.x;
    if (i < n) out[i] = (__bf16)in[i];
}

// ---------------- GEMM1: hidden = relu(x @ W1^T), stored bf16 ----------------
// grid.x = mTiles (N/16), block = 128 (4 waves); wave w owns 64 output columns.
// All 4 waves share the same 16 x-rows -> single HBM read, L2 rebroadcast.
__global__ __launch_bounds__(128) void gemm1_relu(
    const float* __restrict__ x, const __bf16* __restrict__ W1b,
    __bf16* __restrict__ hidden) {
    const int mTile = blockIdx.x;
    const int wave  = threadIdx.x >> 5;
    const int lane  = threadIdx.x & 31;
    const int l     = lane & 15;
    const int hi    = lane >> 4;
    const int nBase = wave * 64;

    const float*  arow = x   + (size_t)(mTile * 16 + l) * N_FEAT + hi * 8;
    const __bf16* brow = W1b + (size_t)(nBase + l)      * N_FEAT + hi * 8;

    v8f acc0 = {}, acc1 = {}, acc2 = {}, acc3 = {};
    for (int kb = 0; kb < N_FEAT / 32; ++kb) {
        const int ko = kb * 32;
        v16bf af = frag_from_f32(arow + ko);
        v16bf b0 = frag_from_bf16(brow + ko);
        v16bf b1 = frag_from_bf16(brow + 16 * N_FEAT + ko);
        v16bf b2 = frag_from_bf16(brow + 32 * N_FEAT + ko);
        v16bf b3 = frag_from_bf16(brow + 48 * N_FEAT + ko);
        acc0 = __builtin_amdgcn_wmma_f32_16x16x32_bf16(false, af, false, b0, (short)0, acc0, false, false);
        acc1 = __builtin_amdgcn_wmma_f32_16x16x32_bf16(false, af, false, b1, (short)0, acc1, false, false);
        acc2 = __builtin_amdgcn_wmma_f32_16x16x32_bf16(false, af, false, b2, (short)0, acc2, false, false);
        acc3 = __builtin_amdgcn_wmma_f32_16x16x32_bf16(false, af, false, b3, (short)0, acc3, false, false);
    }
    const int rowBase = mTile * 16 + hi * 8;
#pragma unroll
    for (int r = 0; r < 8; ++r) {
        size_t ro = (size_t)(rowBase + r) * N_HID + nBase + l;
        hidden[ro]      = (__bf16)fmaxf(acc0[r], 0.0f);
        hidden[ro + 16] = (__bf16)fmaxf(acc1[r], 0.0f);
        hidden[ro + 32] = (__bf16)fmaxf(acc2[r], 0.0f);
        hidden[ro + 48] = (__bf16)fmaxf(acc3[r], 0.0f);
    }
}

// ---------------- GEMM2: h0 = hidden @ W2^T (f32 out) ----------------
__global__ __launch_bounds__(128) void gemm2(
    const __bf16* __restrict__ hidden, const __bf16* __restrict__ W2b,
    float* __restrict__ h0, int mTiles) {
    const int mTile = blockIdx.x * 4 + (threadIdx.x >> 5);
    if (mTile >= mTiles) return;
    const int lane = threadIdx.x & 31;
    const int l    = lane & 15;
    const int hi   = lane >> 4;

    const __bf16* arow = hidden + (size_t)(mTile * 16 + l) * N_HID + hi * 8;
    const __bf16* brow = W2b    + (size_t)l * N_HID + hi * 8;

    v8f acc0 = {}, acc1 = {}, acc2 = {}, acc3 = {};
    for (int kb = 0; kb < N_HID / 32; ++kb) {
        const int ko = kb * 32;
        v16bf af = frag_from_bf16(arow + ko);
        v16bf b0 = frag_from_bf16(brow + ko);
        v16bf b1 = frag_from_bf16(brow + 16 * N_HID + ko);
        v16bf b2 = frag_from_bf16(brow + 32 * N_HID + ko);
        v16bf b3 = frag_from_bf16(brow + 48 * N_HID + ko);
        acc0 = __builtin_amdgcn_wmma_f32_16x16x32_bf16(false, af, false, b0, (short)0, acc0, false, false);
        acc1 = __builtin_amdgcn_wmma_f32_16x16x32_bf16(false, af, false, b1, (short)0, acc1, false, false);
        acc2 = __builtin_amdgcn_wmma_f32_16x16x32_bf16(false, af, false, b2, (short)0, acc2, false, false);
        acc3 = __builtin_amdgcn_wmma_f32_16x16x32_bf16(false, af, false, b3, (short)0, acc3, false, false);
    }
    const int rowBase = mTile * 16 + hi * 8;
#pragma unroll
    for (int r = 0; r < 8; ++r) {
        size_t ro = (size_t)(rowBase + r) * N_CLS + l;
        h0[ro]      = acc0[r];
        h0[ro + 16] = acc1[r];
        h0[ro + 32] = acc2[r];
        h0[ro + 48] = acc3[r];
    }
}

// ---------------- CSR build (once): count, scan, fill ----------------
__global__ void zero_int(int* __restrict__ a, int n) {
    int i = blockIdx.x * blockDim.x + threadIdx.x;
    if (i < n) a[i] = 0;
}
__global__ void count_edges(const int* __restrict__ dst, int* __restrict__ cnt, int e) {
    int i = blockIdx.x * blockDim.x + threadIdx.x;
    if (i < e) atomicAdd(cnt + dst[i], 1);
}
__global__ void dinv_k(const int* __restrict__ cnt, float* __restrict__ dinv, int n) {
    int i = blockIdx.x * blockDim.x + threadIdx.x;
    if (i < n) dinv[i] = rsqrtf((float)cnt[i] + 1.0f);   // +1 self-loop, always > 0
}
// per-block exclusive scan (Hillis-Steele in LDS) + block totals
__global__ __launch_bounds__(256) void scan_block(const int* __restrict__ cnt,
                                                  int* __restrict__ offs,
                                                  int* __restrict__ blockSums, int n) {
    __shared__ int sh[256];
    int i = blockIdx.x * 256 + threadIdx.x;
    int v = (i < n) ? cnt[i] : 0;
    sh[threadIdx.x] = v;
    __syncthreads();
#pragma unroll
    for (int d = 1; d < 256; d <<= 1) {
        int t = (threadIdx.x >= d) ? sh[threadIdx.x - d] : 0;
        __syncthreads();
        sh[threadIdx.x] += t;
        __syncthreads();
    }
    if (i < n) offs[i] = sh[threadIdx.x] - v;            // exclusive within block
    if (threadIdx.x == 255) blockSums[blockIdx.x] = sh[255];
}
__global__ void scan_mid(int* __restrict__ blockSums, int nb) {
    if (blockIdx.x == 0 && threadIdx.x == 0) {
        int run = 0;
        for (int b = 0; b < nb; ++b) { int t = blockSums[b]; blockSums[b] = run; run += t; }
    }
}
__global__ void scan_add(int* __restrict__ offs, const int* __restrict__ blockSums,
                         int* __restrict__ cursor, int n) {
    int i = blockIdx.x * 256 + threadIdx.x;
    if (i < n) { offs[i] += blockSums[i >> 8]; cursor[i] = 0; }
}
// place each edge into its dst bucket: payload = (src, norm) packed in int2
__global__ void fill_csr(const int* __restrict__ src, const int* __restrict__ dst,
                         const float* __restrict__ dinv, const int* __restrict__ offs,
                         int* __restrict__ cursor, int2* __restrict__ perm, int e) {
    int i = blockIdx.x * blockDim.x + threadIdx.x;
    if (i < e) {
        int d = dst[i], s = src[i];
        int p = offs[d] + atomicAdd(cursor + d, 1);
        int2 pk;
        pk.x = s;
        pk.y = __float_as_int(dinv[s] * dinv[d]);
        perm[p] = pk;
    }
}

// ---------------- fused APPNP hop: pull-mode, atomic-free ----------------
// one wave32 per destination node, 2 features per lane (float2), register accum.
// nxt = (1-a) * (dinv^2*cur[node] + sum_e norm_e*cur[src_e]) + a*h0
__global__ __launch_bounds__(256) void appnp_hop(
    const int* __restrict__ offs, const int* __restrict__ cnt,
    const int2* __restrict__ perm, const float* __restrict__ dinv,
    const float* __restrict__ cur, const float* __restrict__ h0,
    float* __restrict__ nxt, int n) {
    int node = blockIdx.x * 8 + (threadIdx.x >> 5);
    if (node >= n) return;
    int lane = threadIdx.x & 31;
    int f = lane << 1;

    float di = dinv[node];
    float2 sv = *(const float2*)(cur + (size_t)node * N_CLS + f);
    float a0 = di * di * sv.x;
    float a1 = di * di * sv.y;

    const int start = offs[node];
    const int deg   = cnt[node];
    const int2* pe  = perm + start;

    int k = 0;
    for (; k + 4 <= deg; k += 4) {
        int2 e0 = pe[k], e1 = pe[k + 1], e2 = pe[k + 2], e3 = pe[k + 3];
        float2 u0 = *(const float2*)(cur + (size_t)e0.x * N_CLS + f);
        float2 u1 = *(const float2*)(cur + (size_t)e1.x * N_CLS + f);
        float2 u2 = *(const float2*)(cur + (size_t)e2.x * N_CLS + f);
        float2 u3 = *(const float2*)(cur + (size_t)e3.x * N_CLS + f);
        float w0 = __int_as_float(e0.y), w1 = __int_as_float(e1.y);
        float w2 = __int_as_float(e2.y), w3 = __int_as_float(e3.y);
        a0 = fmaf(w0, u0.x, a0); a1 = fmaf(w0, u0.y, a1);
        a0 = fmaf(w1, u1.x, a0); a1 = fmaf(w1, u1.y, a1);
        a0 = fmaf(w2, u2.x, a0); a1 = fmaf(w2, u2.y, a1);
        a0 = fmaf(w3, u3.x, a0); a1 = fmaf(w3, u3.y, a1);
    }
    for (; k < deg; ++k) {
        int2 e0 = pe[k];
        float2 u0 = *(const float2*)(cur + (size_t)e0.x * N_CLS + f);
        float w0 = __int_as_float(e0.y);
        a0 = fmaf(w0, u0.x, a0); a1 = fmaf(w0, u0.y, a1);
    }

    float2 h = *(const float2*)(h0 + (size_t)node * N_CLS + f);
    float2 o;
    o.x = (1.0f - ALPHA) * a0 + ALPHA * h.x;
    o.y = (1.0f - ALPHA) * a1 + ALPHA * h.y;
    *(float2*)(nxt + (size_t)node * N_CLS + f) = o;
}

// ---------------- log_softmax: one wave32 per node, 2 classes/lane ----------------
__global__ __launch_bounds__(256) void log_softmax_k(const float* __restrict__ cur,
                                                     float* __restrict__ out, int n) {
    int node = blockIdx.x * 8 + (threadIdx.x >> 5);
    if (node >= n) return;
    int lane = threadIdx.x & 31;
    const float* p = cur + (size_t)node * N_CLS;
    float v0 = p[lane], v1 = p[lane + 32];
    float m = fmaxf(v0, v1);
#pragma unroll
    for (int off = 16; off > 0; off >>= 1) m = fmaxf(m, __shfl_xor(m, off, 32));
    float s = __expf(v0 - m) + __expf(v1 - m);
#pragma unroll
    for (int off = 16; off > 0; off >>= 1) s += __shfl_xor(s, off, 32);
    float lse = m + __logf(s);
    float* q = out + (size_t)node * N_CLS;
    q[lane]      = v0 - lse;
    q[lane + 32] = v1 - lse;
}

// ---------------- launcher ----------------
extern "C" void kernel_launch(void* const* d_in, const int* in_sizes, int n_in,
                              void* d_out, int out_size, void* d_ws, size_t ws_size,
                              hipStream_t stream) {
    const float* x  = (const float*)d_in[0];
    const int*   ei = (const int*)d_in[1];   // [2,E] flat, int per harness contract
    const float* W1 = (const float*)d_in[2];
    const float* W2 = (const float*)d_in[3];

    const int N = in_sizes[0] / N_FEAT;      // 50000
    const int E = in_sizes[1] / 2;           // 1600000
    const int* src = ei;
    const int* dst = ei + E;

    // workspace carve-out (~78 MB total)
    char* w = (char*)d_ws;
    size_t off = 0;
    auto carve = [&](size_t bytes) -> void* {
        void* p = w + off;
        off += (bytes + 255) & ~(size_t)255;
        return p;
    };
    __bf16* W1b    = (__bf16*)carve((size_t)N_HID * N_FEAT * sizeof(__bf16));
    __bf16* W2b    = (__bf16*)carve((size_t)N_CLS * N_HID * sizeof(__bf16));
    __bf16* hidden = (__bf16*)carve((size_t)N * N_HID * sizeof(__bf16));
    float*  h0     = (float*)carve((size_t)N * N_CLS * sizeof(float));
    float*  bufA   = (float*)carve((size_t)N * N_CLS * sizeof(float));
    float*  bufB   = (float*)carve((size_t)N * N_CLS * sizeof(float));
    float*  dinv   = (float*)carve((size_t)N * sizeof(float));
    int*    cnt    = (int*)carve((size_t)N * sizeof(int));
    int*    offs   = (int*)carve((size_t)N * sizeof(int));
    int*    cursor = (int*)carve((size_t)N * sizeof(int));
    int*    bsums  = (int*)carve((size_t)1024 * sizeof(int));
    int2*   perm   = (int2*)carve((size_t)E * sizeof(int2));
    (void)ws_size; (void)n_in; (void)out_size;

    const int mTiles = N / 16;               // 3125
    const int nbN  = (N + 255) / 256;
    const int nbE  = (E + 255) / 256;

    // weights -> bf16 (once per call, tiny)
    cvt_bf16<<<(N_HID * N_FEAT + 255) / 256, 256, 0, stream>>>(W1, W1b, N_HID * N_FEAT);
    cvt_bf16<<<(N_CLS * N_HID + 255) / 256, 256, 0, stream>>>(W2, W2b, N_CLS * N_HID);

    // MLP encoder via WMMA (bf16 in, f32 accum)
    gemm1_relu<<<mTiles, 128, 0, stream>>>(x, W1b, hidden);
    gemm2<<<(mTiles + 3) / 4, 128, 0, stream>>>(hidden, W2b, h0, mTiles);

    // CSR build: count -> dinv -> scan -> fill
    zero_int<<<nbN, 256, 0, stream>>>(cnt, N);
    count_edges<<<nbE, 256, 0, stream>>>(dst, cnt, E);
    dinv_k<<<nbN, 256, 0, stream>>>(cnt, dinv, N);
    scan_block<<<nbN, 256, 0, stream>>>(cnt, offs, bsums, N);
    scan_mid<<<1, 32, 0, stream>>>(bsums, nbN);
    scan_add<<<nbN, 256, 0, stream>>>(offs, bsums, cursor, N);
    fill_csr<<<nbE, 256, 0, stream>>>(src, dst, dinv, offs, cursor, perm, E);

    // APPNP: 10 fused pull-mode hops, ping-pong buffers (all L2-resident)
    const int hopBlocks = (N + 7) / 8;
    const float* rd = h0;
    float* wr = bufA;
    for (int h = 0; h < K_HOPS; ++h) {
        appnp_hop<<<hopBlocks, 256, 0, stream>>>(offs, cnt, perm, dinv, rd, h0, wr, N);
        rd = wr;
        wr = (wr == bufA) ? bufB : bufA;
    }

    // final log-softmax
    log_softmax_k<<<hopBlocks, 256, 0, stream>>>(rd, (float*)d_out, N);
}
